// ActorCritic_89326729822841
// MI455X (gfx1250) — matrix operand order repaired and checked by reading, not compiled, not run on hardware
//
#include <hip/hip_runtime.h>
#include <hip/hip_bf16.h>

// ---------------- CDNA5 WMMA types ----------------
typedef __attribute__((ext_vector_type(16))) __bf16 v16bf;
typedef __attribute__((ext_vector_type(8)))  float  v8f;
typedef __attribute__((ext_vector_type(4)))  int    v4i;

union Frag {
    v16bf v;
    uint4 q[2];
};

// Async global->LDS copy engine (CDNA5). Guarded: falls back to VMEM+convert.
#if defined(__has_builtin)
#  if __has_builtin(__builtin_amdgcn_global_load_async_to_lds_b128)
#    define HAVE_ASYNC 1
#  endif
#endif
#ifndef HAVE_ASYNC
#  define HAVE_ASYNC 0
#endif

#if HAVE_ASYNC
typedef __attribute__((address_space(1))) v4i* gv4i_p;   // global v4i32*
typedef __attribute__((address_space(3))) v4i* lv4i_p;   // LDS v4i32*
#endif

__device__ __forceinline__ unsigned short f2bf(float f) {
    unsigned int u = __float_as_uint(f);
    unsigned int r = u + 0x7FFFu + ((u >> 16) & 1u);   // round-to-nearest-even
    return (unsigned short)(r >> 16);
}

// Problem constants
#define PB  512
#define PN  200
#define PD  256
#define PHC 512
#define PHA 512
#define PA  4096
#define PK  32
#define NROWS (PB * PN)   // 102400 = 3200 tiles of 32

// ---------------------------------------------------------------------------
// Pack fp32 weight matrix W[K x Ncols] (row-major) into BF16 WMMA B-fragments.
// Packed element index p = ((nt*KT + kt)*32 + lane)*16 + j
//   lane 0..15  : N = lane,     K = kt*32 + j       (j = 0..15)
//   lane 16..31 : N = lane-16,  K = kt*32 + 16 + j
// ---------------------------------------------------------------------------
__global__ void pack_w_kernel(const float* __restrict__ W,
                              unsigned short* __restrict__ P,
                              int K, int Ncols) {
    int KT = K >> 5;
    long total = (long)K * Ncols;
    for (long p = (long)blockIdx.x * blockDim.x + threadIdx.x; p < total;
         p += (long)gridDim.x * blockDim.x) {
        int j    = (int)(p & 15);
        int lane = (int)((p >> 4) & 31);
        long rest = p >> 9;
        int kt = (int)(rest % KT);
        long nt = rest / KT;
        int n = lane & 15;
        int k = ((lane >> 4) << 4) + j;
        long row = (long)kt * 32 + k;
        long col = nt * 16 + n;
        P[p] = f2bf(W[row * Ncols + col]);
    }
}

// ---------------------------------------------------------------------------
// Fused critic: node_vs[g][row] = relu(X[row]·Wc1 + bc1) · Wc2 + bc2
// grid = (3200, 2), block = 128 (4 waves). 32-row tile per workgroup,
// 2x M register blocking: each B fragment feeds two WMMAs.
// ---------------------------------------------------------------------------
__global__ __launch_bounds__(128)
void critic_kernel(const float* __restrict__ X0, const float* __restrict__ X1,
                   const unsigned short* __restrict__ Wc1p,
                   const float* __restrict__ bc1, const float* __restrict__ Wc2,
                   const float* __restrict__ bc2, float* __restrict__ node_vs) {
    __shared__ unsigned short As[32 * 264];   // padded stride 264 bf16
    __shared__ float vals[32];
#if HAVE_ASYNC
    __shared__ float Af[32 * 256];            // raw fp32 staging (async engine)
#endif

    const float* X = blockIdx.y ? X1 : X0;
    const long rowbase = (long)blockIdx.x * 32;
    const float* Xblk = X + rowbase * PD;
    const int tid = threadIdx.x;

#if HAVE_ASYNC
    // Stage 32x256 fp32 via GLOBAL_LOAD_ASYNC_TO_LDS_B128 (ASYNCcnt-tracked),
    // then convert to bf16 in LDS. Each thread converts only what it copied,
    // so only the per-wave async wait is needed before the convert.
    for (int e = tid; e < 2048; e += 128) {   // 2048 x 16B
        __builtin_amdgcn_global_load_async_to_lds_b128(
            (gv4i_p)(Xblk + (long)e * 4),
            (lv4i_p)(&Af[e * 4]), 0, 0);
    }
#  if __has_builtin(__builtin_amdgcn_s_wait_asynccnt)
    __builtin_amdgcn_s_wait_asynccnt(0);
#  else
    asm volatile("s_wait_asynccnt 0" ::: "memory");
#  endif
    for (int e = tid; e < 2048; e += 128) {
        int r = e >> 6, c4 = e & 63;
        float4 f = *(const float4*)(&Af[e * 4]);
        int base = r * 264 + c4 * 4;
        *(unsigned int*)(&As[base])     = (unsigned int)f2bf(f.x) | ((unsigned int)f2bf(f.y) << 16);
        *(unsigned int*)(&As[base + 2]) = (unsigned int)f2bf(f.z) | ((unsigned int)f2bf(f.w) << 16);
    }
#else
    for (int e = tid; e < 2048; e += 128) {   // 2048 float4
        int r = e >> 6, c4 = e & 63;
        float4 f = ((const float4*)(Xblk + (long)r * PD))[c4];
        int base = r * 264 + c4 * 4;
        *(unsigned int*)(&As[base])     = (unsigned int)f2bf(f.x) | ((unsigned int)f2bf(f.y) << 16);
        *(unsigned int*)(&As[base + 2]) = (unsigned int)f2bf(f.z) | ((unsigned int)f2bf(f.w) << 16);
    }
#endif
    if (tid < 32) vals[tid] = 0.0f;
    __syncthreads();

    const int lane  = tid & 31;
    const int wave  = tid >> 5;
    const int m16   = lane & 15;
    const int hi16  = lane >> 4;
    const int kbase = hi16 * 16;

    float part0[8], part1[8];
#pragma unroll
    for (int v = 0; v < 8; ++v) { part0[v] = 0.0f; part1[v] = 0.0f; }

    for (int nti = 0; nti < 8; ++nti) {
        const int nt = wave * 8 + nti;
        v8f acc0 = {}, acc1 = {};
#pragma unroll
        for (int kt = 0; kt < 8; ++kt) {
            Frag a0, a1, b;
            a0.q[0] = *(const uint4*)(&As[m16 * 264 + kt * 32 + kbase]);
            a0.q[1] = *(const uint4*)(&As[m16 * 264 + kt * 32 + kbase + 8]);
            a1.q[0] = *(const uint4*)(&As[(m16 + 16) * 264 + kt * 32 + kbase]);
            a1.q[1] = *(const uint4*)(&As[(m16 + 16) * 264 + kt * 32 + kbase + 8]);
            const unsigned short* bp = Wc1p + ((long)(nt * 8 + kt) * 512) + lane * 16;
            b.q[0] = ((const uint4*)bp)[0];
            b.q[1] = ((const uint4*)bp)[1];
            if (kt < 7) __builtin_prefetch(bp + 512, 0, 1);
            acc0 = __builtin_amdgcn_wmma_f32_16x16x32_bf16(
                false, a0.v, false, b.v, (short)0, acc0, false, false);
            acc1 = __builtin_amdgcn_wmma_f32_16x16x32_bf16(
                false, a1.v, false, b.v, (short)0, acc1, false, false);
        }
        const int n = nt * 16 + m16;
        const float bias = bc1[n];
        const float w2   = Wc2[n];
#pragma unroll
        for (int v = 0; v < 8; ++v) {
            float h0 = acc0[v] + bias;
            float h1 = acc1[v] + bias;
            h0 = h0 > 0.0f ? h0 : 0.0f;
            h1 = h1 > 0.0f ? h1 : 0.0f;
            part0[v] += h0 * w2;
            part1[v] += h1 * w2;
        }
    }
    // Reduce across the 16 N-lanes (xor masks stay within each 16-lane half)
#pragma unroll
    for (int off = 1; off < 16; off <<= 1)
#pragma unroll
        for (int v = 0; v < 8; ++v) {
            part0[v] += __shfl_xor(part0[v], off, 32);
            part1[v] += __shfl_xor(part1[v], off, 32);
        }
    if (m16 == 0) {
#pragma unroll
        for (int v = 0; v < 8; ++v) {
            atomicAdd(&vals[v + hi16 * 8], part0[v]);        // ds_add_f32
            atomicAdd(&vals[16 + v + hi16 * 8], part1[v]);
        }
    }
    __syncthreads();
    if (tid < 32)
        node_vs[(long)blockIdx.y * NROWS + rowbase + tid] = vals[tid] + bc2[0];
}

// ---------------------------------------------------------------------------
// Actor GEMM1: gather selected rows, H1 = bf16(relu(sel·Wa1 + ba1))
// grid = 32, block = 128
// ---------------------------------------------------------------------------
__global__ __launch_bounds__(128)
void actor1_kernel(const float* __restrict__ X, const int* __restrict__ nodes,
                   const unsigned short* __restrict__ Wa1p,
                   const float* __restrict__ ba1, unsigned short* __restrict__ H1) {
    __shared__ unsigned short As[16 * 264];
    __shared__ long rowsrc[16];
    const int tid = threadIdx.x;
    const int tile = blockIdx.x;

    if (tid < 16) {
        int b = tile * 16 + tid;
        rowsrc[tid] = (long)b * PN + nodes[b];
    }
    __syncthreads();
    for (int e = tid; e < 1024; e += 128) {
        int r = e >> 6, c4 = e & 63;
        float4 f = ((const float4*)(X + rowsrc[r] * PD))[c4];
        int base = r * 264 + c4 * 4;
        *(unsigned int*)(&As[base])     = (unsigned int)f2bf(f.x) | ((unsigned int)f2bf(f.y) << 16);
        *(unsigned int*)(&As[base + 2]) = (unsigned int)f2bf(f.z) | ((unsigned int)f2bf(f.w) << 16);
    }
    __syncthreads();

    const int lane = tid & 31, wave = tid >> 5;
    const int m16 = lane & 15, hi16 = lane >> 4, kbase = hi16 * 16;

    for (int nti = 0; nti < 8; ++nti) {
        const int nt = wave * 8 + nti;
        v8f acc = {};
#pragma unroll
        for (int kt = 0; kt < 8; ++kt) {
            Frag a, b;
            a.q[0] = *(const uint4*)(&As[m16 * 264 + kt * 32 + kbase]);
            a.q[1] = *(const uint4*)(&As[m16 * 264 + kt * 32 + kbase + 8]);
            const unsigned short* bp = Wa1p + ((long)(nt * 8 + kt) * 512) + lane * 16;
            b.q[0] = ((const uint4*)bp)[0];
            b.q[1] = ((const uint4*)bp)[1];
            acc = __builtin_amdgcn_wmma_f32_16x16x32_bf16(
                false, a.v, false, b.v, (short)0, acc, false, false);
        }
        const int n = nt * 16 + m16;
        const float bias = ba1[n];
#pragma unroll
        for (int v = 0; v < 8; ++v) {
            float h = acc[v] + bias;
            h = h > 0.0f ? h : 0.0f;
            H1[(long)(tile * 16 + v + hi16 * 8) * PHA + n] = f2bf(h);
        }
    }
}

// ---------------------------------------------------------------------------
// Actor GEMM2: logits = H1(bf16)·Wa2 + ba2   grid = (32, 8), block = 128
// ---------------------------------------------------------------------------
__global__ __launch_bounds__(128)
void actor2_kernel(const unsigned short* __restrict__ H1,
                   const unsigned short* __restrict__ Wa2p,
                   const float* __restrict__ ba2, float* __restrict__ logits) {
    __shared__ unsigned short As[16 * 520];   // 16 rows x 512 bf16, padded
    const int tid = threadIdx.x;
    const int tile = blockIdx.x;
    const int nchunk = blockIdx.y;

    for (int e = tid; e < 1024; e += 128) {   // 1024 uint4 = 16KB
        int r = e >> 5, c = e & 31;
        ((uint4*)(&As[r * 520]))[c] =
            ((const uint4*)(H1 + (long)(tile * 16 + r) * PHA))[c];
    }
    __syncthreads();

    const int lane = tid & 31, wave = tid >> 5;
    const int m16 = lane & 15, hi16 = lane >> 4, kbase = hi16 * 16;

    for (int nti = 0; nti < 8; ++nti) {
        const int nt = nchunk * 32 + wave * 8 + nti;
        v8f acc = {};
#pragma unroll
        for (int kt = 0; kt < 16; ++kt) {
            Frag a, b;
            a.q[0] = *(const uint4*)(&As[m16 * 520 + kt * 32 + kbase]);
            a.q[1] = *(const uint4*)(&As[m16 * 520 + kt * 32 + kbase + 8]);
            const unsigned short* bp = Wa2p + ((long)(nt * 16 + kt) * 512) + lane * 16;
            b.q[0] = ((const uint4*)bp)[0];
            b.q[1] = ((const uint4*)bp)[1];
            if (kt < 15) __builtin_prefetch(bp + 512, 0, 1);
            acc = __builtin_amdgcn_wmma_f32_16x16x32_bf16(
                false, a.v, false, b.v, (short)0, acc, false, false);
        }
        const int n = nt * 16 + m16;
        const float bias = ba2[n];
#pragma unroll
        for (int v = 0; v < 8; ++v)
            logits[(long)(tile * 16 + v + hi16 * 8) * PA + n] = acc[v] + bias;
    }
}

// ---------------------------------------------------------------------------
// Masked log-softmax / entropy / value gathers. grid = 512, block = 256.
// ---------------------------------------------------------------------------
__global__ __launch_bounds__(256)
void softmax_kernel(const float* __restrict__ logits,
                    const unsigned char* __restrict__ masks,
                    const int* __restrict__ xfers, const int* __restrict__ nodes,
                    const int* __restrict__ nnl,
                    const unsigned char* __restrict__ is_term,
                    const float* __restrict__ node_vs,
                    float* __restrict__ ent_ws, float* __restrict__ out) {
    __shared__ float red[256];
    const int b = blockIdx.x;
    const int tid = threadIdx.x;
    const float* L = logits + (long)b * PA;
    const unsigned char* Mk = masks + (long)b * PA;

    float mx = -3.4e38f;
    for (int a = tid; a < PA; a += 256) {
        float ml = L[a] - 1e10f * (1.0f - (float)Mk[a]);
        mx = fmaxf(mx, ml);
    }
    red[tid] = mx; __syncthreads();
    for (int s = 128; s > 0; s >>= 1) {
        if (tid < s) red[tid] = fmaxf(red[tid], red[tid + s]);
        __syncthreads();
    }
    mx = red[0]; __syncthreads();

    float s1 = 0.0f, s2 = 0.0f;
    for (int a = tid; a < PA; a += 256) {
        float ml = L[a] - 1e10f * (1.0f - (float)Mk[a]);
        float e = expf(ml - mx);
        s1 += e; s2 += e * ml;
    }
    red[tid] = s1; __syncthreads();
    for (int s = 128; s > 0; s >>= 1) {
        if (tid < s) red[tid] += red[tid + s];
        __syncthreads();
    }
    s1 = red[0]; __syncthreads();
    red[tid] = s2; __syncthreads();
    for (int s = 128; s > 0; s >>= 1) {
        if (tid < s) red[tid] += red[tid + s];
        __syncthreads();
    }
    s2 = red[0];

    if (tid == 0) {
        float logZ = mx + logf(s1);
        int xf = xfers[b];
        float mlx = L[xf] - 1e10f * (1.0f - (float)Mk[xf]);
        out[2 * PB + b] = mlx - logZ;          // xfer_logprobs
        ent_ws[b] = logZ - s2 / s1;            // per-row entropy
        out[b] = node_vs[(long)b * PN + nodes[b]];  // values
        float mxv = -3.4e38f;
        for (int k = 0; k < PK; ++k) {
            int nn = nnl[b * PK + k];
            mxv = fmaxf(mxv, node_vs[(long)NROWS + (long)b * PN + nn]);
        }
        out[PB + b] = is_term[b] ? 0.0f : mxv; // next_values
    }
}

__global__ __launch_bounds__(256)
void entropy_kernel(const float* __restrict__ ent_ws, float* __restrict__ out) {
    __shared__ float red[256];
    const int tid = threadIdx.x;
    red[tid] = ent_ws[tid] + ent_ws[tid + 256];
    __syncthreads();
    for (int s = 128; s > 0; s >>= 1) {
        if (tid < s) red[tid] += red[tid + s];
        __syncthreads();
    }
    if (tid == 0) out[3 * PB] = red[0] / (float)PB;
}

// ---------------------------------------------------------------------------
extern "C" void kernel_launch(void* const* d_in, const int* in_sizes, int n_in,
                              void* d_out, int out_size, void* d_ws, size_t ws_size,
                              hipStream_t stream) {
    const float* graph_embeds      = (const float*)d_in[0];
    const float* next_graph_embeds = (const float*)d_in[1];
    const float* Wc1 = (const float*)d_in[2];
    const float* bc1 = (const float*)d_in[3];
    const float* Wc2 = (const float*)d_in[4];
    const float* bc2 = (const float*)d_in[5];
    const float* Wa1 = (const float*)d_in[6];
    const float* ba1 = (const float*)d_in[7];
    const float* Wa2 = (const float*)d_in[8];
    const float* ba2 = (const float*)d_in[9];
    const int* nodes = (const int*)d_in[10];
    const int* xfers = (const int*)d_in[11];
    const int* nnl   = (const int*)d_in[12];
    const unsigned char* is_term = (const unsigned char*)d_in[13];
    const unsigned char* masks   = (const unsigned char*)d_in[14];
    float* out = (float*)d_out;

    char* ws = (char*)d_ws;
    unsigned short* Wc1p = (unsigned short*)(ws);                 // 256KB
    unsigned short* Wa1p = (unsigned short*)(ws + 262144);        // 256KB
    unsigned short* Wa2p = (unsigned short*)(ws + 524288);        // 4MB
    float* node_vs       = (float*)(ws + 4718592);                // 800KB (2x102400)
    unsigned short* H1   = (unsigned short*)(ws + 5537792);       // 512KB
    float* logits        = (float*)(ws + 6062080);                // 8MB
    float* ent_ws        = (float*)(ws + 14450688);               // 2KB

    hipLaunchKernelGGL(pack_w_kernel, dim3(512), dim3(256), 0, stream, Wc1, Wc1p, PD, PHC);
    hipLaunchKernelGGL(pack_w_kernel, dim3(512), dim3(256), 0, stream, Wa1, Wa1p, PD, PHA);
    hipLaunchKernelGGL(pack_w_kernel, dim3(2048), dim3(256), 0, stream, Wa2, Wa2p, PHA, PA);

    hipLaunchKernelGGL(critic_kernel, dim3(NROWS / 32, 2), dim3(128), 0, stream,
                       graph_embeds, next_graph_embeds, Wc1p, bc1, Wc2, bc2, node_vs);

    hipLaunchKernelGGL(actor1_kernel, dim3(PB / 16), dim3(128), 0, stream,
                       graph_embeds, nodes, Wa1p, ba1, H1);

    hipLaunchKernelGGL(actor2_kernel, dim3(PB / 16, 8), dim3(128), 0, stream,
                       H1, Wa2p, ba2, logits);

    hipLaunchKernelGGL(softmax_kernel, dim3(PB), dim3(256), 0, stream,
                       logits, masks, xfers, nodes, nnl, is_term, node_vs, ent_ws, out);

    hipLaunchKernelGGL(entropy_kernel, dim3(1), dim3(256), 0, stream, ent_ws, out);
}
